// SparseNGCNLayer_59090160058611
// MI455X (gfx1250) — matrix operand order repaired and checked by reading, not compiled, not run on hardware
//
#include <hip/hip_runtime.h>

typedef __attribute__((ext_vector_type(16))) __bf16 v16bf;
typedef __attribute__((ext_vector_type(8)))  __bf16 v8bf;   // 16 bytes
typedef __attribute__((ext_vector_type(8)))  float  v8f;
typedef __attribute__((ext_vector_type(4)))  float  f4;
typedef __attribute__((ext_vector_type(2)))  float  f2;

#define N_NODES 50000
#define N_EDGES 800000
#define IN_CH   512
#define OUT_CH  64
#define BM      128             // rows per workgroup (8 wave32 tiles of 16)
#define BK      32              // K per step (16x16x32 bf16 WMMA)
#define NSTEP   (IN_CH / BK)    // 16

// ---------------------------------------------------------------------------
// Dense transform + ReLU:  out[N,64] = relu(feat[N,512] @ W[512,64])
// bf16 WMMA, f32 accumulate. Double-buffered LDS staging in ISA fragment
// layout. Prefetch for step s+1 is issued AFTER the barrier so the
// pre-barrier wait is dscnt-only (no loadcnt stall on fresh prefetches).
// ---------------------------------------------------------------------------
__global__ __launch_bounds__(256)
void gemm_relu_bf16_wmma(const float* __restrict__ feat,
                         const float* __restrict__ Wm,
                         float* __restrict__ out)
{
    // A fragments: 2 bufs x 8 m-tiles x 32 lanes x 16 bf16  (16 KB)
    // B fragments: 2 bufs x 4 n-tiles x 32 lanes x 16 bf16  ( 8 KB)
    __shared__ __align__(32) __bf16 sA[2][8 * 32 * 16];
    __shared__ __align__(32) __bf16 sB[2][4 * 32 * 16];

    const int tid  = threadIdx.x;
    const int wave = tid >> 5;           // 0..7 -> m-tile
    const int lane = tid & 31;
    const int rowBase = blockIdx.x * BM;

    // ---- A staging geometry: thread owns (m, k in [kbase, kbase+16))
    const int am    = tid >> 1;                       // 0..127
    const int akb   = (tid & 1) * 16;                 // 0 or 16
    const int arow  = min(rowBase + am, N_NODES - 1); // clamp: tail rows never stored
    const int atile = am >> 4;
    const int amm   = am & 15;
    const int ael0  = akb >> 1;                       // element offset 0 or 8
    const float* agp = feat + (long)arow * IN_CH + akb;

    // ---- B staging geometry: thread owns (kk, n in [n0, n0+8))
    const int bkk   = tid >> 3;                       // 0..31 (K row within step)
    const int bn0   = (tid & 7) * 8;                  // 0..56
    const int btile = bn0 >> 4;
    const int bel0  = bn0 & 15;                       // 0 or 8
    const float* bgp = Wm + bn0;                      // + (k0+bkk)*64 per step

    v8f acc0 = {}, acc1 = {}, acc2 = {}, acc3 = {};

    // prologue: global loads for step 0
    f4 ra0 = ((const f4*)agp)[0];
    f4 ra1 = ((const f4*)agp)[1];
    f4 ra2 = ((const f4*)agp)[2];
    f4 ra3 = ((const f4*)agp)[3];
    f4 rb0 = ((const f4*)(bgp + (long)bkk * OUT_CH))[0];
    f4 rb1 = ((const f4*)(bgp + (long)bkk * OUT_CH))[1];

    int buf = 0;
    for (int s = 0; s < NSTEP; ++s, buf ^= 1) {
        // ---- convert + store fragments (2x ds_store_b128 for A, 8x ds_store_b16 for B)
        v8bf c0, c1, cb;
        #pragma unroll
        for (int u = 0; u < 4; ++u) {
            c0[u]     = (__bf16)ra0[u];  c0[4 + u] = (__bf16)ra1[u];
            c1[u]     = (__bf16)ra2[u];  c1[4 + u] = (__bf16)ra3[u];
            cb[u]     = (__bf16)rb0[u];  cb[4 + u] = (__bf16)rb1[u];
        }
        *(v8bf*)&sA[buf][(atile * 32 + amm     ) * 16 + ael0] = c0;
        *(v8bf*)&sA[buf][(atile * 32 + amm + 16) * 16 + ael0] = c1;
        {
            const int bh = (bkk >> 3) & 1;
            const int bs = (bkk & 7) + ((bkk >> 4) << 3);
            #pragma unroll
            for (int j = 0; j < 8; ++j)
                sB[buf][(btile * 32 + bel0 + j + 16 * bh) * 16 + bs] = cb[j];
        }

        __syncthreads();

        // ---- issue next step's global loads AFTER the barrier: they overlap
        // the fragment ds_loads + WMMAs and aren't drained by the barrier wait
        const int sn = (s + 1 < NSTEP) ? (s + 1) : (NSTEP - 1);
        const float* agn = agp + sn * BK;
        const float* bgn = bgp + (long)(sn * BK + bkk) * OUT_CH;
        ra0 = ((const f4*)agn)[0];
        ra1 = ((const f4*)agn)[1];
        ra2 = ((const f4*)agn)[2];
        ra3 = ((const f4*)agn)[3];
        rb0 = ((const f4*)bgn)[0];
        rb1 = ((const f4*)bgn)[1];

        // ---- fragments + 4 WMMAs (N = 4 tiles of 16)
        const v16bf a  = *(const v16bf*)(&sA[buf][(wave * 32 + lane) * 16]);
        const v16bf b0 = *(const v16bf*)(&sB[buf][(0 * 32 + lane) * 16]);
        const v16bf b1 = *(const v16bf*)(&sB[buf][(1 * 32 + lane) * 16]);
        const v16bf b2 = *(const v16bf*)(&sB[buf][(2 * 32 + lane) * 16]);
        const v16bf b3 = *(const v16bf*)(&sB[buf][(3 * 32 + lane) * 16]);

        acc0 = __builtin_amdgcn_wmma_f32_16x16x32_bf16(false, a, false, b0, (short)0, acc0, false, false);
        acc1 = __builtin_amdgcn_wmma_f32_16x16x32_bf16(false, a, false, b1, (short)0, acc1, false, false);
        acc2 = __builtin_amdgcn_wmma_f32_16x16x32_bf16(false, a, false, b2, (short)0, acc2, false, false);
        acc3 = __builtin_amdgcn_wmma_f32_16x16x32_bf16(false, a, false, b3, (short)0, acc3, false, false);
    }

    // ---- ReLU + store. C layout: VGPR r, lane L -> M = r + 8*(L/16), N = L%16
    const int nLane = lane & 15;
    const int mHalf = (lane >> 4) * 8;
    #pragma unroll
    for (int r = 0; r < 8; ++r) {
        const int row = rowBase + wave * 16 + mHalf + r;
        if (row < N_NODES) {
            float* op = out + (long)row * OUT_CH + nLane;
            float v0 = acc0[r]; op[ 0] = v0 > 0.0f ? v0 : 0.0f;
            float v1 = acc1[r]; op[16] = v1 > 0.0f ? v1 : 0.0f;
            float v2 = acc2[r]; op[32] = v2 > 0.0f ? v2 : 0.0f;
            float v3 = acc3[r]; op[48] = v3 > 0.0f ? v3 : 0.0f;
        }
    }
}

// ---------------------------------------------------------------------------
// Sparse propagation: out[dst[e], :] += w[e] * in[src[e], :]
// One wave32 per edge, 2 channels per thread: float2 gathers + 2 hardware
// f32 atomics (f32 atomics are 1 dword each -- this is the minimum count).
// Edge index is wave-uniform -> readfirstlane -> scalar metadata loads.
// N_EDGES*32 is an exact multiple of 256 -> fully branchless.
// ---------------------------------------------------------------------------
__global__ __launch_bounds__(256)
void scatter_edges(const int* __restrict__ esrc,
                   const int* __restrict__ edst,
                   const float* __restrict__ ew,
                   const float* __restrict__ in,
                   float* __restrict__ out)
{
    const long idx  = (long)blockIdx.x * blockDim.x + threadIdx.x;
    const int  lane = (int)(idx & 31);
    const int  e    = __builtin_amdgcn_readfirstlane((int)(idx >> 5));

    const int   s = esrc[e];
    const int   d = edst[e];
    const float w = ew[e];

    const f2 x = ((const f2*)(in + (long)s * OUT_CH))[lane];
    float* op  = out + (long)d * OUT_CH + 2 * lane;
    unsafeAtomicAdd(op,     w * x.x);
    unsafeAtomicAdd(op + 1, w * x.y);
}

__global__ __launch_bounds__(256)
void zero_f32v(f4* __restrict__ p)   // exact grid: no bounds check needed
{
    const int i = blockIdx.x * blockDim.x + threadIdx.x;
    p[i] = (f4){0.0f, 0.0f, 0.0f, 0.0f};
}

// ---------------------------------------------------------------------------
extern "C" void kernel_launch(void* const* d_in, const int* in_sizes, int n_in,
                              void* d_out, int out_size, void* d_ws, size_t ws_size,
                              hipStream_t stream)
{
    const int*   edge_src = (const int*)d_in[0];
    const int*   edge_dst = (const int*)d_in[1];
    const float* edge_w   = (const float*)d_in[2];
    const float* feat     = (const float*)d_in[3];
    const float* Wm       = (const float*)d_in[4];

    float* out = (float*)d_out;
    float* ws0 = (float*)d_ws;                 // 50000*64*4 = 12.8 MB ping buffer

    const int total      = N_NODES * OUT_CH;               // 3,200,000 (mult of 1024)
    const int zeroBlocks = total / 4 / 256;                // 3125, exact
    const int gemmBlocks = (N_NODES + BM - 1) / BM;        // 391
    const int scatBlocks = (int)(((long)N_EDGES * 32) / 256);  // 100000, exact

    // base = relu(feat @ W)  -> ws0
    gemm_relu_bf16_wmma<<<gemmBlocks, 256, 0, stream>>>(feat, Wm, ws0);

    // pass 1: ws0 -> out
    zero_f32v<<<zeroBlocks, 256, 0, stream>>>((f4*)out);
    scatter_edges<<<scatBlocks, 256, 0, stream>>>(edge_src, edge_dst, edge_w, ws0, out);

    // pass 2: out -> ws0
    zero_f32v<<<zeroBlocks, 256, 0, stream>>>((f4*)ws0);
    scatter_edges<<<scatBlocks, 256, 0, stream>>>(edge_src, edge_dst, edge_w, out, ws0);

    // final result back to d_out (graph-capture-safe d2d copy)
    hipMemcpyAsync(out, ws0, (size_t)total * sizeof(float),
                   hipMemcpyDeviceToDevice, stream);
}
